// MoE_KAN_MLP_12421045420327
// MI455X (gfx1250) — matrix operand order repaired and checked by reading, not compiled, not run on hardware
//
#include <hip/hip_runtime.h>

// ---------------------------------------------------------------------------
// MoE (4 MLP + 4 FasterKAN experts), full-capacity dense, bf16 WMMA compute.
//   T=2048 tokens, H=1024, F=4096, F2=2048, G=8 grids, top-2 router.
// GEMM: 128x128 tile, KB=64, double-buffered LDS fed by
//       global_load_async_to_lds_b128 (ASYNCcnt) overlapping the WMMAs.
// Workspace layout (needs ws_size >= ~76 MB):
//   wfull 64KB | xb 4MB | hbuf 4MB | actA 16MB | zbuf 16MB | wt 34MB
// ---------------------------------------------------------------------------

typedef __attribute__((ext_vector_type(16))) __bf16 bf16x16;
typedef __attribute__((ext_vector_type(8)))  float  floatx8;

struct FragU { union { bf16x16 v; uint4 q[2]; }; };

__device__ __forceinline__ float b2f(unsigned short u) {
    return __uint_as_float(((unsigned)u) << 16);
}
__device__ __forceinline__ unsigned short f2bf(float f) {
    unsigned u = __float_as_uint(f);
    u += 0x7fffu + ((u >> 16) & 1u);           // round-to-nearest-even
    return (unsigned short)(u >> 16);
}

// Async 64B/thread copy global -> LDS. The ISA applies INST_OFFSET to BOTH
// the LDS and global addresses (CDNA5 ISA), so one
// base pair + offset:16/32/48 walks both sides together. Tracked by ASYNCcnt.
__device__ __forceinline__ void async_copy64(unsigned lds, const void* g) {
    asm volatile(
        "global_load_async_to_lds_b128 %0, %1, off\n\t"
        "global_load_async_to_lds_b128 %0, %1, off offset:16\n\t"
        "global_load_async_to_lds_b128 %0, %1, off offset:32\n\t"
        "global_load_async_to_lds_b128 %0, %1, off offset:48"
        :: "v"(lds), "v"(g) : "memory");
}
__device__ __forceinline__ void wait_async0() {
    asm volatile("s_wait_asynccnt 0x0" ::: "memory");
}

// RSWAF basis: k = h*8+g, basis = sech^2((v-grid[g])*0.5), grid = -1.2+0.2g
__device__ __forceinline__ void expand_store_rswaf(uint2 hu,
                                                   unsigned short* dst) {
    union { uint2 u; unsigned short h[4]; } hv; hv.u = hu;
    union { unsigned short s[32]; uint4 q[4]; } pk;
    #pragma unroll
    for (int j = 0; j < 4; ++j) {
        const float v = b2f(hv.h[j]);
        #pragma unroll
        for (int g = 0; g < 8; ++g) {
            const float d = fabsf(v - (-1.2f + 0.2f * (float)g));
            const float s = __expf(-d);          // v_exp_f32: co-execs w/ WMMA
            const float o = 1.0f + s;
            pk.s[j * 8 + g] = f2bf(4.0f * s / (o * o));
        }
    }
    uint4* d4 = (uint4*)dst;
    d4[0] = pk.q[0]; d4[1] = pk.q[1]; d4[2] = pk.q[2]; d4[3] = pk.q[3];
}

// --------------------------- small utility kernels -------------------------

__global__ void zero_kernel(float* __restrict__ p, int n) {
    int i = blockIdx.x * 256 + threadIdx.x;
    if (i < n) p[i] = 0.0f;
}

__global__ void cvt_bf16_kernel(const float* __restrict__ in,
                                unsigned short* __restrict__ out, int n) {
    int i = blockIdx.x * 256 + threadIdx.x;
    if (i < n) out[i] = f2bf(in[i]);
}

// Router: logits = x @ gate_W, fp32 softmax, top-2, renormalize -> w_full[T,8]
__global__ __launch_bounds__(128) void router_kernel(
    const float* __restrict__ x, const float* __restrict__ gw,
    float* __restrict__ wfull, int H) {
    const int t = blockIdx.x;
    __shared__ float red[128][8];
    float acc[8] = {0,0,0,0,0,0,0,0};
    for (int h = threadIdx.x; h < H; h += 128) {
        float xv = x[(size_t)t * H + h];
        const float4* g4 = (const float4*)(gw + (size_t)h * 8);
        float4 ga = g4[0], gb = g4[1];
        acc[0] += xv * ga.x; acc[1] += xv * ga.y;
        acc[2] += xv * ga.z; acc[3] += xv * ga.w;
        acc[4] += xv * gb.x; acc[5] += xv * gb.y;
        acc[6] += xv * gb.z; acc[7] += xv * gb.w;
    }
    #pragma unroll
    for (int e = 0; e < 8; ++e) red[threadIdx.x][e] = acc[e];
    __syncthreads();
    for (int s = 64; s > 0; s >>= 1) {
        if (threadIdx.x < (unsigned)s)
            #pragma unroll
            for (int e = 0; e < 8; ++e)
                red[threadIdx.x][e] += red[threadIdx.x + s][e];
        __syncthreads();
    }
    if (threadIdx.x == 0) {
        float lg[8], p[8];
        float mx = -3.4e38f;
        #pragma unroll
        for (int e = 0; e < 8; ++e) { lg[e] = red[0][e]; mx = fmaxf(mx, lg[e]); }
        float sum = 0.0f;
        #pragma unroll
        for (int e = 0; e < 8; ++e) { p[e] = __expf(lg[e] - mx); sum += p[e]; }
        float inv = 1.0f / sum;
        #pragma unroll
        for (int e = 0; e < 8; ++e) p[e] *= inv;
        int i1 = 0;
        #pragma unroll
        for (int e = 1; e < 8; ++e) if (p[e] > p[i1]) i1 = e;
        int i2 = (i1 == 0) ? 1 : 0;
        #pragma unroll
        for (int e = 0; e < 8; ++e) if (e != i1 && p[e] > p[i2]) i2 = e;
        float norm = 1.0f / (p[i1] + p[i2]);
        #pragma unroll
        for (int e = 0; e < 8; ++e) {
            float w = (e == i1) ? p[i1] * norm : ((e == i2) ? p[i2] * norm : 0.0f);
            wfull[(size_t)t * 8 + e] = w;
        }
    }
}

// LayerNorm (fp32 in -> bf16 out), one block per token
__global__ __launch_bounds__(256) void ln_kernel(
    const float* __restrict__ in, const float* __restrict__ g,
    const float* __restrict__ b, unsigned short* __restrict__ out, int W) {
    const int t = blockIdx.x;
    const float* row = in + (size_t)t * W;
    float s = 0.0f, ss = 0.0f;
    for (int i = threadIdx.x; i < W; i += 256) {
        float v = row[i]; s += v; ss += v * v;
    }
    __shared__ float rs[256], rq[256];
    rs[threadIdx.x] = s; rq[threadIdx.x] = ss;
    __syncthreads();
    for (int st = 128; st > 0; st >>= 1) {
        if (threadIdx.x < (unsigned)st) {
            rs[threadIdx.x] += rs[threadIdx.x + st];
            rq[threadIdx.x] += rq[threadIdx.x + st];
        }
        __syncthreads();
    }
    __shared__ float mu_s, rstd_s;
    if (threadIdx.x == 0) {
        float mu  = rs[0] / (float)W;
        float var = rq[0] / (float)W - mu * mu;
        mu_s = mu; rstd_s = rsqrtf(var + 1e-5f);
    }
    __syncthreads();
    const float mu = mu_s, rstd = rstd_s;
    for (int i = threadIdx.x; i < W; i += 256)
        out[(size_t)t * W + i] = f2bf((row[i] - mu) * rstd * g[i] + b[i]);
}

// Weight convert + transpose: fp32 [K][N] -> bf16 [N][K]
__global__ __launch_bounds__(256) void wcvt_kernel(
    const float* __restrict__ W, unsigned short* __restrict__ out, int K, int N) {
    __shared__ float tile[32][33];
    const int tx = threadIdx.x & 31, ty = threadIdx.x >> 5;   // ty: 0..7
    const int n0 = blockIdx.x * 32, k0 = blockIdx.y * 32;
    #pragma unroll
    for (int i = 0; i < 4; ++i)
        tile[ty + 8 * i][tx] = W[(size_t)(k0 + ty + 8 * i) * N + n0 + tx];
    __syncthreads();
    #pragma unroll
    for (int i = 0; i < 4; ++i)
        out[(size_t)(n0 + ty + 8 * i) * K + k0 + tx] = f2bf(tile[tx][ty + 8 * i]);
}

// --------------------------- WMMA GEMM ------------------------------------
// C[M,N] = A[M,K] (bf16, or RSWAF-expanded from A[M,K/8]) @ Bt[N,K] (bf16)
// EPI: 0 = +bias, exact GELU, store bf16   (MLP layer 1)
//      1 = +bias, store fp32               (KAN layer 1 -> zbuf)
//      2 = d_out[m,n] += wfull[m,expert]*(c+bias)  (final combine)
template<int EPI, bool EXPAND>
__global__ __launch_bounds__(256) void gemm_wmma_kernel(
    const unsigned short* __restrict__ A,
    const unsigned short* __restrict__ Bt,
    const float* __restrict__ bias,
    const float* __restrict__ wfull, int expert,
    float* __restrict__ outF, unsigned short* __restrict__ outH,
    int M, int N, int K) {
    constexpr int LDP = 72;   // row pad: stride 144 B -> conflict-free frags
    __shared__ unsigned short As[2][128][LDP];   // double-buffered
    __shared__ unsigned short Bs[2][128][LDP];

    const int tid  = threadIdx.x;
    const int m0   = blockIdx.y * 128;
    const int n0   = blockIdx.x * 128;
    const int lane = tid & 31;
    const int wv   = tid >> 5;
    const int wm   = wv >> 1;          // 0..3  (M direction)
    const int wn   = wv & 1;           // 0..1  (N direction)
    const int lrow = lane & 15;
    const int lhi  = lane >> 4;        // half-wave selector per ISA layout

    const floatx8 zero8 = {0,0,0,0,0,0,0,0};
    floatx8 acc[2][4];
    #pragma unroll
    for (int mt = 0; mt < 2; ++mt)
        #pragma unroll
        for (int nt = 0; nt < 4; ++nt) acc[mt][nt] = zero8;

    const int copyRow  = tid >> 1;
    const int copyHalf = tid & 1;
    const int kTiles   = K >> 6;       // KB = 64

    // per-thread staging addresses (64B/thread per tile per stage)
    const unsigned short* bsrc = Bt + (size_t)(n0 + copyRow) * K + copyHalf * 32;
    const unsigned ldsB[2] = {
        (unsigned)(uintptr_t)&Bs[0][copyRow][copyHalf * 32],
        (unsigned)(uintptr_t)&Bs[1][copyRow][copyHalf * 32] };
    const unsigned ldsA[2] = {
        (unsigned)(uintptr_t)&As[0][copyRow][copyHalf * 32],
        (unsigned)(uintptr_t)&As[1][copyRow][copyHalf * 32] };
    const unsigned short* asrc = nullptr;
    const unsigned short* hrow = nullptr;
    uint2 hA = make_uint2(0u, 0u);     // h-values for stage kt+1 (EXPAND)

    if constexpr (EXPAND) {
        const int Hin = K >> 3;
        hrow = A + (size_t)(m0 + copyRow) * Hin + copyHalf * 4;
    } else {
        asrc = A + (size_t)(m0 + copyRow) * K + copyHalf * 32;
    }

    // ---- prologue: stage 0 (and preload h for stage 1) ----
    if constexpr (EXPAND) {
        uint2 h0 = *(const uint2*)hrow;
        expand_store_rswaf(h0, &As[0][copyRow][copyHalf * 32]);
        if (kTiles > 1) hA = *(const uint2*)(hrow + 8);
    } else {
        async_copy64(ldsA[0], asrc);
    }
    async_copy64(ldsB[0], bsrc);

    for (int kt = 0; kt < kTiles; ++kt) {
        wait_async0();                 // my async stores for buf kt&1 landed
        __syncthreads();               // everyone's stage kt ready, kt-1 read
        if (kt + 1 < kTiles) {
            const int nb = (kt + 1) & 1;
            if constexpr (EXPAND) {
                expand_store_rswaf(hA, &As[nb][copyRow][copyHalf * 32]);
                if (kt + 2 < kTiles)
                    hA = *(const uint2*)(hrow + (size_t)(kt + 2) * 8);
            } else {
                async_copy64(ldsA[nb], asrc + (size_t)(kt + 1) * 64);
            }
            async_copy64(ldsB[nb], bsrc + (size_t)(kt + 1) * 64);
            __builtin_prefetch(bsrc + (size_t)(kt + 3) * 64, 0, 1);
        }
        const int cb = kt & 1;
        // ---- 2 WMMA K-steps of 32: preload all frags, then 8 WMMAs ----
        #pragma unroll
        for (int kk = 0; kk < 64; kk += 32) {
            FragU af[2], bfr[4];
            #pragma unroll
            for (int mt = 0; mt < 2; ++mt) {
                // A 16x32 bf16: lane holds K runs [sel*8,+8) and [16+sel*8,+8)
                const unsigned short* pa =
                    &As[cb][wm * 32 + mt * 16 + lrow][kk + lhi * 8];
                af[mt].q[0] = *(const uint4*)pa;
                af[mt].q[1] = *(const uint4*)(pa + 16);
            }
            #pragma unroll
            for (int nt = 0; nt < 4; ++nt) {
                // B 32x16 bf16: lane holds K runs [sel*8,+8) and [16+sel*8,+8) at col lane&15
                const unsigned short* pb =
                    &Bs[cb][wn * 64 + nt * 16 + lrow][kk + lhi * 8];
                bfr[nt].q[0] = *(const uint4*)pb;
                bfr[nt].q[1] = *(const uint4*)(pb + 16);
            }
            #pragma unroll
            for (int nt = 0; nt < 4; ++nt)
                #pragma unroll
                for (int mt = 0; mt < 2; ++mt)
                    acc[mt][nt] = __builtin_amdgcn_wmma_f32_16x16x32_bf16(
                        false, af[mt].v, false, bfr[nt].v,
                        (short)0, acc[mt][nt], false, false);
        }
    }

    // ---- epilogue (C layout: VGPR j -> M=j(+8 for lanes 16-31), N=lane&15) --
    #pragma unroll
    for (int mt = 0; mt < 2; ++mt) {
        #pragma unroll
        for (int nt = 0; nt < 4; ++nt) {
            const int n = n0 + wn * 64 + nt * 16 + lrow;
            const float bv = bias[n];
            #pragma unroll
            for (int j = 0; j < 8; ++j) {
                const int m = m0 + wm * 32 + mt * 16 + lhi * 8 + j;
                float c = acc[mt][nt][j] + bv;
                if constexpr (EPI == 0) {
                    c = 0.5f * c * (1.0f + erff(c * 0.70710678118654752f));
                    outH[(size_t)m * N + n] = f2bf(c);
                } else if constexpr (EPI == 1) {
                    outF[(size_t)m * N + n] = c;
                } else {
                    const float w = wfull[(size_t)m * 8 + expert];
                    outF[(size_t)m * N + n] += w * c;
                }
            }
        }
    }
}

// --------------------------- host-side launcher ----------------------------

extern "C" void kernel_launch(void* const* d_in, const int* in_sizes, int n_in,
                              void* d_out, int out_size, void* d_ws, size_t ws_size,
                              hipStream_t stream) {
    constexpr int H = 1024, F = 4096, F2 = 2048, G = 8;
    const float* x      = (const float*)d_in[0];
    const float* gateW  = (const float*)d_in[1];
    const float* mlp_W1 = (const float*)d_in[2];
    const float* mlp_b1 = (const float*)d_in[3];
    const float* mlp_W2 = (const float*)d_in[4];
    const float* mlp_b2 = (const float*)d_in[5];
    const float* g0     = (const float*)d_in[6];
    const float* b0     = (const float*)d_in[7];
    const float* kW0    = (const float*)d_in[8];
    const float* kb0    = (const float*)d_in[9];
    const float* g1     = (const float*)d_in[10];
    const float* b1     = (const float*)d_in[11];
    const float* kW1    = (const float*)d_in[12];
    const float* kb1    = (const float*)d_in[13];
    float* out = (float*)d_out;
    const int T = in_sizes[0] / H;   // 2048 (multiple of 128)

    // workspace layout (bytes, 256-aligned)
    char* ws = (char*)d_ws;
    float*          wfull = (float*)(ws);                           // 64 KB
    unsigned short* xb    = (unsigned short*)(ws + (1u << 20));     // 4 MB
    unsigned short* hbuf  = (unsigned short*)(ws + (5u << 20));     // 4 MB
    unsigned short* actA  = (unsigned short*)(ws + (9u << 20));     // 16 MB
    float*          zbuf  = (float*)(ws + (25u << 20));             // 16 MB
    unsigned short* wt    = (unsigned short*)(ws + (41u << 20));    // 34 MB

    const int TH = T * H;
    zero_kernel<<<(TH + 255) / 256, 256, 0, stream>>>(out, TH);
    router_kernel<<<T, 128, 0, stream>>>(x, gateW, wfull, H);
    cvt_bf16_kernel<<<(TH + 255) / 256, 256, 0, stream>>>(x, xb, TH);

    // ---- MLP experts 0..3: Linear(H,F) -> GELU -> Linear(F,H) ----
    for (int e = 0; e < 4; ++e) {
        wcvt_kernel<<<dim3(F / 32, H / 32), 256, 0, stream>>>(
            mlp_W1 + (size_t)e * H * F, wt, H, F);
        gemm_wmma_kernel<0, false><<<dim3(F / 128, T / 128), 256, 0, stream>>>(
            xb, wt, mlp_b1 + (size_t)e * F, nullptr, 0, nullptr, actA, T, F, H);
        wcvt_kernel<<<dim3(H / 32, F / 32), 256, 0, stream>>>(
            mlp_W2 + (size_t)e * F * H, wt, F, H);
        gemm_wmma_kernel<2, false><<<dim3(H / 128, T / 128), 256, 0, stream>>>(
            actA, wt, mlp_b2 + (size_t)e * H, wfull, e, out, nullptr, T, H, F);
    }

    // ---- KAN experts 4..7: (LN -> RSWAF -> SplineLinear) x 2 ----
    for (int e = 0; e < 4; ++e) {
        ln_kernel<<<T, 256, 0, stream>>>(
            x, g0 + (size_t)e * H, b0 + (size_t)e * H, hbuf, H);
        wcvt_kernel<<<dim3(F2 / 32, (H * G) / 32), 256, 0, stream>>>(
            kW0 + (size_t)e * H * G * F2, wt, H * G, F2);
        gemm_wmma_kernel<1, true><<<dim3(F2 / 128, T / 128), 256, 0, stream>>>(
            hbuf, wt, kb0 + (size_t)e * F2, nullptr, 0, zbuf, nullptr,
            T, F2, H * G);
        ln_kernel<<<T, 256, 0, stream>>>(
            zbuf, g1 + (size_t)e * F2, b1 + (size_t)e * F2, actA, F2);
        wcvt_kernel<<<dim3(H / 32, (F2 * G) / 32), 256, 0, stream>>>(
            kW1 + (size_t)e * F2 * G * H, wt, F2 * G, H);
        gemm_wmma_kernel<2, true><<<dim3(H / 128, T / 128), 256, 0, stream>>>(
            actA, wt, kb1 + (size_t)e * H, wfull, 4 + e, out, nullptr,
            T, H, F2 * G);
    }
}
